// TopKSAE_50483045598043
// MI455X (gfx1250) — matrix operand order repaired and checked by reading, not compiled, not run on hardware
//
#include <hip/hip_runtime.h>
#include <hip/hip_bf16.h>
#include <stdint.h>

#define D_MODEL 1024
#define N_FEAT  16384
#define BATCH   8192
#define TOPK    64

#define KC      64            // K per staged chunk (2 WMMA k-steps)
#define NCHUNK  (D_MODEL / KC)
#define ROWP    144           // padded LDS row pitch: 128B data + 16B pad
#define ABUF    (64 * ROWP)   // one A buffer: 64 rows

typedef __attribute__((ext_vector_type(16))) __bf16 v16bf;
typedef __attribute__((ext_vector_type(8)))  float  v8f;
typedef __attribute__((ext_vector_type(4)))  unsigned int v4u;
typedef __attribute__((ext_vector_type(8)))  int v8i;
typedef __attribute__((ext_vector_type(4)))  int v4i;

union FragU { uint4 q[2]; v16bf v; };

#if defined(__AMDGCN__) && __has_builtin(__builtin_amdgcn_tensor_load_to_lds) && \
    __has_builtin(__builtin_amdgcn_s_wait_tensorcnt)
#define USE_TDM 1
#endif

__device__ __forceinline__ unsigned short f32_to_bf16_rne(float f) {
  unsigned u = __float_as_uint(f);
  u += 0x7FFFu + ((u >> 16) & 1u);
  return (unsigned short)(u >> 16);
}

// monotonic float<->u32 key (ascending)
__device__ __forceinline__ unsigned sortable_from_bits(unsigned u) {
  return u ^ ((u & 0x80000000u) ? 0xFFFFFFFFu : 0x80000000u);
}
__device__ __forceinline__ unsigned bits_from_sortable(unsigned k) {
  return k ^ ((k & 0x80000000u) ? 0x80000000u : 0xFFFFFFFFu);
}

// ---------------- prep kernels: fold b_dec into x, convert to bf16 ----------
__global__ void k_prep_x(const float* __restrict__ x, const float* __restrict__ b_dec,
                         unsigned short* __restrict__ xc) {
  for (int i = blockIdx.x * blockDim.x + threadIdx.x; i < BATCH * D_MODEL;
       i += gridDim.x * blockDim.x)
    xc[i] = f32_to_bf16_rne(x[i] - b_dec[i & (D_MODEL - 1)]);
}

__global__ void k_prep_w(const float* __restrict__ w, unsigned short* __restrict__ wc, int n) {
  for (int i = blockIdx.x * blockDim.x + threadIdx.x; i < n; i += gridDim.x * blockDim.x)
    wc[i] = f32_to_bf16_rne(w[i]);
}

#ifdef USE_TDM
// Issue a TDM 2-D tile load: 64 rows x 64 K bf16 from row-major [*,1024] into
// LDS with 16B padding per 128B row (ROWP=144).  Uniform args; wave-scalar op.
__device__ __forceinline__ void tdm_issue_a(const unsigned short* gsrc, unsigned ldsOff) {
  unsigned long long ga = (unsigned long long)(uintptr_t)gsrc;
  // D# group0: count=1, lds_addr, global_addr[56:0], type=2
  v4u g0 = { 1u,
             ldsOff,
             (unsigned)ga,
             (unsigned)((ga >> 32) & 0x01FFFFFFu) | (2u << 30) };
  // D# group1: data_size=2B, pad_enable, pad_interval=32 DW, pad_amount=4 DW,
  // tensor_dim0=1024, tensor_dim1=64, tile_dim0=64, tile_dim1=64,
  // tensor_dim0_stride=1024
  v8i g1 = { (int)((1u << 16) | (1u << 20) | (4u << 22) | (3u << 25)),
             (int)(1024u << 16),   // tensor_dim0[15:0] in [31:16]
             (int)(64u << 16),     // tensor_dim0 hi=0 | tensor_dim1[15:0]
             (int)(64u << 16),     // tensor_dim1 hi=0 | tile_dim0
             (int)64,              // tile_dim1 | tile_dim2=0
             (int)1024,            // tensor_dim0_stride lo
             0, 0 };
  v4i z4 = {0, 0, 0, 0};
#if __clang_major__ >= 23
  v8i z8 = {0, 0, 0, 0, 0, 0, 0, 0};
  __builtin_amdgcn_tensor_load_to_lds(g0, g1, z4, z4, z8, 0);
#else
  __builtin_amdgcn_tensor_load_to_lds(g0, g1, z4, z4, 0);
#endif
}
#endif

// ---------------- encoder GEMM: pre = xc @ Wenc^T + b_enc -------------------
// Block = 256 threads = 8 waves arranged 2(M) x 4(N).  Block tile 64M x 128N.
// Wave tile 32M x 32N via 4 accumulators.  A (64 rows x 64K) staged in LDS by
// the Tensor Data Mover (double-buffered, s_wait_tensorcnt), B streamed from
// global (hits L2: whole W_enc bf16 = 32 MB < 192 MB L2).
__global__ void __launch_bounds__(256)
k_enc_gemm(const unsigned short* __restrict__ xc,
           const unsigned short* __restrict__ wenc,
           const float* __restrict__ b_enc,
           float* __restrict__ pre) {
  __shared__ __align__(16) unsigned char As[2 * ABUF];

  const int lane  = threadIdx.x & 31;
  const int wave  = threadIdx.x >> 5;
  const int waveM = wave >> 2;          // 0..1
  const int waveN = wave & 3;           // 0..3

  const int mBlock = blockIdx.y * 64;
  const int nW     = blockIdx.x * 128 + waveN * 32;

  const int r16   = lane & 15;
  const int hi    = lane >> 4;          // 0 or 1
  const int bKoff = hi << 3;            // B: K {8h..+7} and {16+8h..+7} per half-wave

  const unsigned short* bPtr0 = wenc + (size_t)(nW + r16) * D_MODEL;
  const unsigned short* bPtr1 = bPtr0 + (size_t)16 * D_MODEL;
  const unsigned short* aG    = xc + (size_t)mBlock * D_MODEL;

  // A fragment base inside padded LDS tile: row = waveM*32 + r16 (+16 for the
  // second M half), byte offset hi*16 selects K 0-7 vs 8-15 within a k-step.
  const int aLds = (waveM * 32 + r16) * ROWP + hi * 16;

  v8f c00 = {}, c01 = {}, c10 = {}, c11 = {};

#ifdef USE_TDM
  if (wave == 0) tdm_issue_a(aG, (unsigned)(size_t)(As));
#endif

  for (int ci = 0; ci < NCHUNK; ++ci) {
    const int bufOff = (ci & 1) * ABUF;
#ifdef USE_TDM
    if (wave == 0) __builtin_amdgcn_s_wait_tensorcnt(0);  // chunk ci landed
    __syncthreads();
    if (wave == 0 && ci + 1 < NCHUNK)                     // overlap next DMA
      tdm_issue_a(aG + (ci + 1) * KC, (unsigned)(size_t)(As + (((ci + 1) & 1) * ABUF)));
#else
    __syncthreads();
    for (int i = threadIdx.x; i < 512; i += 256) {        // cooperative copy
      const int r = i >> 3, seg = i & 7;
      *(uint4*)(As + bufOff + r * ROWP + seg * 16) =
          *(const uint4*)(aG + (size_t)r * D_MODEL + ci * KC + seg * 8);
    }
    __syncthreads();
#endif

    if (ci + 1 < NCHUNK) {
      __builtin_prefetch(bPtr0 + (ci + 1) * KC, 0, 1);    // global_prefetch_b8
      __builtin_prefetch(bPtr1 + (ci + 1) * KC, 0, 1);
    }

#pragma unroll
    for (int s = 0; s < 2; ++s) {                         // two 32-K WMMA steps
      const int kb = ci * KC + s * 32;
      FragU a0, a1, b0, b1;
      const uint4* ap0 = (const uint4*)(As + bufOff + aLds + s * 64);
      const uint4* ap1 = (const uint4*)(As + bufOff + aLds + 16 * ROWP + s * 64);
      a0.q[0] = ap0[0]; a0.q[1] = ap0[2];   // K {koff..+7} and {16+koff..+7}
      a1.q[0] = ap1[0]; a1.q[1] = ap1[2];
      const uint4* pb0 = (const uint4*)(bPtr0 + kb + bKoff);
      const uint4* pb1 = (const uint4*)(bPtr1 + kb + bKoff);
      b0.q[0] = pb0[0]; b0.q[1] = pb0[2];
      b1.q[0] = pb1[0]; b1.q[1] = pb1[2];

      c00 = __builtin_amdgcn_wmma_f32_16x16x32_bf16(false, a0.v, false, b0.v,
                                                    (short)0, c00, false, false);
      c01 = __builtin_amdgcn_wmma_f32_16x16x32_bf16(false, a0.v, false, b1.v,
                                                    (short)0, c01, false, false);
      c10 = __builtin_amdgcn_wmma_f32_16x16x32_bf16(false, a1.v, false, b0.v,
                                                    (short)0, c10, false, false);
      c11 = __builtin_amdgcn_wmma_f32_16x16x32_bf16(false, a1.v, false, b1.v,
                                                    (short)0, c11, false, false);
    }
  }

  // C/D layout: VGPR g -> M = g + 8*hi ; N = lane%16.
  const int mW   = mBlock + waveM * 32;
  const int n0   = nW + r16;
  const int n1   = n0 + 16;
  const float bi0 = b_enc[n0];
  const float bi1 = b_enc[n1];
  const int rowSel = hi << 3;
#pragma unroll
  for (int g = 0; g < 8; ++g) {
    const int r0 = mW + g + rowSel;
    pre[(size_t)r0 * N_FEAT + n0]        = c00[g] + bi0;
    pre[(size_t)r0 * N_FEAT + n1]        = c01[g] + bi1;
    pre[(size_t)(r0 + 16) * N_FEAT + n0] = c10[g] + bi0;
    pre[(size_t)(r0 + 16) * N_FEAT + n1] = c11[g] + bi1;
  }
}

// ---------------- per-row exact top-64 + decode + loss ----------------------
// Single-pass 4096-bin (top-12-bit) histogram select, one collect pass,
// 256-wide bitonic sort (value desc, index asc on ties), fused decode + MSE.
__global__ void __launch_bounds__(256)
k_topk_decode(const float* __restrict__ pre,
              const float* __restrict__ x,
              const float* __restrict__ W_dec,
              const float* __restrict__ b_dec,
              float* __restrict__ xhat,
              float* __restrict__ vals_out,
              float* __restrict__ idx_out,
              float* __restrict__ lossAcc) {
  __shared__ int hist[4096];
  __shared__ int sup[64];
  __shared__ int sh_selBin;
  __shared__ int sh_cnt;
  __shared__ unsigned long long keyBuf[256];
  __shared__ float s_val[TOPK];
  __shared__ int   s_idx[TOPK];
  __shared__ float red[256];

  const int row = blockIdx.x;
  const int tid = threadIdx.x;
  const float* rowPtr = pre + (size_t)row * N_FEAT;

  for (int i = tid; i < 4096; i += 256) hist[i] = 0;
  keyBuf[tid] = 0ull;
  if (tid == 0) sh_cnt = 0;
  __syncthreads();

  // scan 1: 12-bit histogram
  for (int j = tid; j < N_FEAT; j += 256) {
    unsigned key = sortable_from_bits(__float_as_uint(rowPtr[j]));
    atomicAdd(&hist[key >> 20], 1);
  }
  __syncthreads();

  // find the bin containing the 64th-largest key (walk from the top)
  if (tid < 64) {
    int s = 0;
#pragma unroll 8
    for (int b = 0; b < 64; ++b) s += hist[tid * 64 + b];
    sup[tid] = s;
  }
  __syncthreads();
  if (tid == 0) {
    int cum = 0, sb = 0;
    for (int sI = 63; sI >= 0; --sI) {
      if (cum + sup[sI] >= TOPK) {
        for (int b = 63; b >= 0; --b) {
          int c = hist[sI * 64 + b];
          if (cum + c >= TOPK) { sb = sI * 64 + b; break; }
          cum += c;
        }
        break;
      }
      cum += sup[sI];
    }
    sh_selBin = sb;
  }
  __syncthreads();
  const unsigned selBin = (unsigned)sh_selBin;

  // scan 2: collect every element with bin >= selBin (contains the top 64)
  for (int j = tid; j < N_FEAT; j += 256) {
    unsigned key = sortable_from_bits(__float_as_uint(rowPtr[j]));
    if ((key >> 20) >= selBin) {
      int p = atomicAdd(&sh_cnt, 1);
      if (p < 256)
        keyBuf[p] = ((unsigned long long)key << 32) | (unsigned)(~(unsigned)j);
    }
  }
  __syncthreads();

  // bitonic sort 256 descending; ties -> larger ~idx first == smaller index
  for (int size = 2; size <= 256; size <<= 1) {
    for (int stride = size >> 1; stride > 0; stride >>= 1) {
      __syncthreads();
      const int j = tid ^ stride;
      if (j > tid) {
        unsigned long long a = keyBuf[tid], b = keyBuf[j];
        const bool doSwap = ((tid & size) == 0) ? (a < b) : (a > b);
        if (doSwap) { keyBuf[tid] = b; keyBuf[j] = a; }
      }
    }
  }
  __syncthreads();

  if (tid < TOPK) {
    unsigned long long bk = keyBuf[tid];
    int   idx = (int)(~(unsigned)(bk & 0xFFFFFFFFull));
    float v   = __uint_as_float(bits_from_sortable((unsigned)(bk >> 32)));
    v = fmaxf(v, 0.0f);                       // ReLU
    s_val[tid] = v;
    s_idx[tid] = idx;
    vals_out[(size_t)row * TOPK + tid] = v;
    idx_out [(size_t)row * TOPK + tid] = (float)idx;
  }
  __syncthreads();

  // decode: x_hat = sum_k v_k * W_dec[idx_k] + b_dec, plus squared error
  float sq = 0.0f;
#pragma unroll
  for (int cc = 0; cc < D_MODEL / 256; ++cc) {
    const int col = tid + cc * 256;
    float acc = b_dec[col];
#pragma unroll 8
    for (int k = 0; k < TOPK; ++k)
      acc = fmaf(s_val[k], W_dec[(size_t)s_idx[k] * D_MODEL + col], acc);
    xhat[(size_t)row * D_MODEL + col] = acc;
    float d = acc - x[(size_t)row * D_MODEL + col];
    sq = fmaf(d, d, sq);
  }
  red[tid] = sq;
  __syncthreads();
  for (int s = 128; s > 0; s >>= 1) {
    if (tid < s) red[tid] += red[tid + s];
    __syncthreads();
  }
  if (tid == 0) atomicAdd(lossAcc, red[0]);
}

__global__ void k_zero_loss(float* lossAcc) { *lossAcc = 0.0f; }

__global__ void k_finalize(const float* __restrict__ lossAcc, float* __restrict__ scalars) {
  float recon = *lossAcc / (float)((long long)BATCH * (long long)D_MODEL);
  scalars[0] = recon;   // recon_loss
  scalars[1] = 0.0f;    // aux_loss
  scalars[2] = recon;   // loss
}

extern "C" void kernel_launch(void* const* d_in, const int* in_sizes, int n_in,
                              void* d_out, int out_size, void* d_ws, size_t ws_size,
                              hipStream_t stream) {
  const float* x     = (const float*)d_in[0];
  const float* W_enc = (const float*)d_in[1];
  const float* b_enc = (const float*)d_in[2];
  const float* W_dec = (const float*)d_in[3];
  const float* b_dec = (const float*)d_in[4];

  // Workspace layout (~587 MB): xc bf16 16MB | wenc bf16 32MB | pre f32 512MB | loss 4B
  char* ws = (char*)d_ws;
  unsigned short* xc   = (unsigned short*)ws;
  unsigned short* wenc = (unsigned short*)(ws + (size_t)BATCH * D_MODEL * 2);
  float* pre = (float*)(ws + (size_t)BATCH * D_MODEL * 2 + (size_t)N_FEAT * D_MODEL * 2);
  float* lossAcc = (float*)(ws + (size_t)BATCH * D_MODEL * 2 +
                            (size_t)N_FEAT * D_MODEL * 2 +
                            (size_t)BATCH * N_FEAT * 4);

  float* out     = (float*)d_out;
  float* xhat    = out;
  float* vals    = out + (size_t)BATCH * D_MODEL;
  float* idxo    = vals + (size_t)BATCH * TOPK;
  float* scalars = idxo + (size_t)BATCH * TOPK;

  k_prep_x<<<2048, 256, 0, stream>>>(x, b_dec, xc);
  k_prep_w<<<4096, 256, 0, stream>>>(W_enc, wenc, N_FEAT * D_MODEL);

  dim3 g(N_FEAT / 128, BATCH / 64);
  k_enc_gemm<<<g, 256, 0, stream>>>(xc, wenc, b_enc, pre);

  k_zero_loss<<<1, 1, 0, stream>>>(lossAcc);
  k_topk_decode<<<BATCH, 256, 0, stream>>>(pre, x, W_dec, b_dec,
                                           xhat, vals, idxo, lossAcc);
  k_finalize<<<1, 1, 0, stream>>>(lossAcc, scalars);
}
